// GCN_Graph_87780541595739
// MI455X (gfx1250) — matrix operand order repaired and checked by reading, not compiled, not run on hardware
//
#include <hip/hip_runtime.h>

typedef __attribute__((ext_vector_type(16))) __bf16 v16bf;
typedef __attribute__((ext_vector_type(8)))  __bf16 v8bf;
typedef __attribute__((ext_vector_type(4)))  __bf16 v4bf;
typedef __attribute__((ext_vector_type(8)))  float  v8f;

#define HD 256          // hidden dim
#define EPS 1e-5f

__device__ __forceinline__ float sigmoidf_(float x){ return 1.f/(1.f+__expf(-x)); }
__device__ __forceinline__ unsigned f2ord(float f){
  unsigned u=__float_as_uint(f); return (u&0x80000000u)? ~u : (u|0x80000000u);
}
__device__ __forceinline__ float ord2f(unsigned u){
  return (u&0x80000000u)? __uint_as_float(u&0x7fffffffu) : __uint_as_float(~u);
}

// ---------------- fills / converts ----------------
__global__ void fill_f32(float* p, float v, int n){
  int i = blockIdx.x*blockDim.x + threadIdx.x;
  if (i<n) p[i]=v;
}
__global__ void fill_u32(unsigned* p, unsigned v, int n){
  int i = blockIdx.x*blockDim.x + threadIdx.x;
  if (i<n) p[i]=v;
}
// 4-wide fp32 -> bf16 (all our sizes are multiples of 4)
__global__ void cvt_bf16x4_kernel(const float* __restrict__ src, __bf16* __restrict__ dst, int n4){
  int i = blockIdx.x*blockDim.x + threadIdx.x;
  if (i<n4){
    float4 v = ((const float4*)src)[i];
    v4bf o = { (__bf16)v.x, (__bf16)v.y, (__bf16)v.z, (__bf16)v.w };
    ((v4bf*)dst)[i] = o;
  }
}

// ---------------- atom encoder ----------------
__device__ const int ATOM_OFF[9] = {0,119,123,135,147,157,163,169,171};

__global__ void embed_kernel(const int* __restrict__ x, const float* __restrict__ emb,
                             float* __restrict__ h, int N){
  int n = blockIdx.x; int t = threadIdx.x;    // 256 threads = feature dim
  float s = 0.f;
  #pragma unroll
  for (int c=0;c<9;++c){
    int row = x[n*9+c] + ATOM_OFF[c];
    s += emb[(size_t)row*HD + t];
  }
  h[(size_t)n*HD + t] = s;
}

// ---------------- degrees ----------------
__global__ void degree_kernel(const int* __restrict__ ei, float* __restrict__ deg, int E){
  int e = blockIdx.x*blockDim.x + threadIdx.x;
  if (e<E) atomicAdd(&deg[ei[E+e]], 1.0f);
}
__global__ void dinv_kernel(float* deg, int N){
  int n = blockIdx.x*blockDim.x + threadIdx.x;
  if (n<N) deg[n] = rsqrtf(deg[n] + 1.0f);
}

// ---------------- WMMA GEMM:  C[M,Nc] (+)= A[M,KD] @ Wb[Nc,KD]^T (+ bias) ----------------
// one wave per block; wave computes a 16 x (NT*16) tile.
// A and Wb pre-converted bf16; KD is compile-time so the fully-unrolled K loop
// uses pure base+immediate addressing (no per-tile 64-bit address math, no spills).
template<int NT, int KD>
__global__ void gemm_wmma(const __bf16* __restrict__ A, const __bf16* __restrict__ Wb,
                          const float* __restrict__ bias, float* __restrict__ C,
                          int M, int Nc, int accumulate){
  const int lane = threadIdx.x;           // 0..31
  const int half = lane >> 4;             // 0 | 1
  const int l16  = lane & 15;
  const int col0 = blockIdx.y*(NT*16);

  v8f acc[NT];
  #pragma unroll
  for (int t=0;t<NT;++t) acc[t] = (v8f){0,0,0,0,0,0,0,0};

  // A frag K split per lane-half: {0..7,16..23} / {8..15,24..31}  (ISA 7.12.2)
  const __bf16* arow  = A  + (size_t)(blockIdx.x*16 + l16)*KD + half*8;
  // B frag K split per lane-half: {0..7,16..23} / {8..15,24..31} (same as A)
  const __bf16* wbase = Wb + (size_t)(col0 + l16)*KD + half*8;

  #pragma unroll
  for (int kk = 0; kk < KD; kk += 32){
    v8bf alo = *(const v8bf*)(arow + kk);
    v8bf ahi = *(const v8bf*)(arow + kk + 16);
    v16bf afrag = __builtin_shufflevector(alo, ahi,
                    0,1,2,3,4,5,6,7,8,9,10,11,12,13,14,15);
    #pragma unroll
    for (int t=0;t<NT;++t){
      v8bf blo = *(const v8bf*)(wbase + (size_t)(t*16*KD + kk));
      v8bf bhi = *(const v8bf*)(wbase + (size_t)(t*16*KD + kk + 16));
      v16bf bfrag = __builtin_shufflevector(blo, bhi,
                    0,1,2,3,4,5,6,7,8,9,10,11,12,13,14,15);
      acc[t] = __builtin_amdgcn_wmma_f32_16x16x32_bf16(false, afrag, false, bfrag,
                                                       (short)0, acc[t], false, false);
    }
  }

  // ---- store (C/D 16x16 f32 layout: VGPR v -> M = v + 8*half, N = l16) ----
  #pragma unroll
  for (int t=0;t<NT;++t){
    int col = col0 + t*16 + l16;
    float bv = bias ? bias[col] : 0.f;
    #pragma unroll
    for (int v=0; v<8; ++v){
      int row = blockIdx.x*16 + v + 8*half;
      size_t idx = (size_t)row*Nc + col;
      float val = acc[t][v] + bv;
      if (accumulate) val += C[idx];
      C[idx] = val;
    }
  }
}

// ---------------- GCN aggregation ----------------
// agg = hw * dinv^2 + b   (self-loop + bias init)
__global__ void agg_init_kernel(const float* __restrict__ hw, const float* __restrict__ dinv,
                                const float* __restrict__ b, float* __restrict__ agg, int N){
  int n = blockIdx.x; int t = threadIdx.x;
  float d = dinv[n];
  agg[(size_t)n*HD+t] = hw[(size_t)n*HD+t]*d*d + b[t];
}
// agg[tgt] += hw[src] * dinv[src]*dinv[tgt]
__global__ void edge_scatter_kernel(const int* __restrict__ ei, const float* __restrict__ hw,
                                    const float* __restrict__ dinv, float* __restrict__ agg, int E){
  int e = blockIdx.x; int t = threadIdx.x;
  int s = ei[e], tg = ei[E+e];
  float w = dinv[s]*dinv[tg];
  atomicAdd(&agg[(size_t)tg*HD+t], hw[(size_t)s*HD+t]*w);
}

// relu in place + per-feature sum / sumsq (block = 256 threads, 256 rows per block)
__global__ void relu_stats_kernel(float* __restrict__ h, float* __restrict__ stats, int N){
  int t = threadIdx.x;
  size_t row0 = (size_t)blockIdx.x * 256;
  float s=0.f, ss=0.f;
  for (int r=0;r<256;++r){
    size_t row = row0 + r;
    if (row < (size_t)N){
      float v = h[row*HD + t];
      v = fmaxf(v, 0.f);
      h[row*HD + t] = v;
      s += v; ss += v*v;
    }
  }
  atomicAdd(&stats[t], s);
  atomicAdd(&stats[HD+t], ss);
}

__global__ void bn_apply_kernel(float* __restrict__ h, const float* __restrict__ stats,
                                const float* __restrict__ gamma, const float* __restrict__ beta,
                                int N){
  int n = blockIdx.x; int t = threadIdx.x;
  float inv = 1.f/(float)N;
  float mu  = stats[t]*inv;
  float var = stats[HD+t]*inv - mu*mu;
  float sc  = rsqrtf(var + EPS)*gamma[t];
  size_t idx = (size_t)n*HD + t;
  h[idx] = (h[idx]-mu)*sc + beta[t];
}

// ---------------- set2set ----------------
__global__ void lstm_kernel(const float* __restrict__ gates, float* __restrict__ hs,
                            float* __restrict__ cs, int total){
  int i = blockIdx.x*blockDim.x + threadIdx.x;
  if (i>=total) return;
  int g = i / HD, t = i % HD;
  const float* gp = gates + (size_t)g*4*HD;
  float ig = sigmoidf_(gp[t]);
  float fg = sigmoidf_(gp[HD+t]);
  float gg = tanhf(gp[2*HD+t]);
  float og = sigmoidf_(gp[3*HD+t]);
  float c = fg*cs[i] + ig*gg;
  cs[i] = c;
  hs[i] = og*tanhf(c);
}

// e[n] = dot(h[n], hs[batch[n]]); also segment-max via ordered-uint atomicMax
__global__ void e_dot_kernel(const float* __restrict__ h, const float* __restrict__ hs,
                             const int* __restrict__ batch, float* __restrict__ e_arr,
                             unsigned* __restrict__ m_arr, int N){
  int node = blockIdx.x*(blockDim.x>>5) + (threadIdx.x>>5);
  int lane = threadIdx.x & 31;
  if (node >= N) return;
  int g = batch[node];
  const float* hp = h  + (size_t)node*HD;
  const float* qp = hs + (size_t)g*HD;
  float s = 0.f;
  for (int i=lane; i<HD; i+=32) s += hp[i]*qp[i];
  #pragma unroll
  for (int off=16; off; off>>=1) s += __shfl_down(s, off, 32);
  if (lane==0){ e_arr[node]=s; atomicMax(&m_arr[g], f2ord(s)); }
}

__global__ void zexp_kernel(float* __restrict__ e_arr, const unsigned* __restrict__ m_arr,
                            float* __restrict__ z, const int* __restrict__ batch, int N){
  int n = blockIdx.x*blockDim.x + threadIdx.x;
  if (n>=N) return;
  int g = batch[n];
  float ee = __expf(e_arr[n] - ord2f(m_arr[g]));
  e_arr[n] = ee;
  atomicAdd(&z[g], ee);
}

__global__ void r_scatter_kernel(const float* __restrict__ e_arr, const float* __restrict__ z,
                                 const float* __restrict__ h, const int* __restrict__ batch,
                                 float* __restrict__ r){
  int n = blockIdx.x; int t = threadIdx.x;
  int g = batch[n];
  float a = e_arr[n] / z[g];
  atomicAdd(&r[(size_t)g*HD+t], a*h[(size_t)n*HD+t]);
}

__global__ void qstar_update_kernel(const float* __restrict__ hs, const float* __restrict__ r,
                                    float* __restrict__ qstar, int total){
  int i = blockIdx.x*blockDim.x + threadIdx.x;
  if (i>=total) return;
  int g = i / HD, t = i % HD;
  qstar[(size_t)g*2*HD + t]      = hs[i];
  qstar[(size_t)g*2*HD + HD + t] = r[i];
}

// ---------------- final lin2 + sigmoid ----------------
__global__ void final_kernel(const float* __restrict__ out1, const float* __restrict__ l2W,
                             const float* __restrict__ l2b, float* __restrict__ out,
                             int G, int ENC){
  int gid  = blockIdx.x*(blockDim.x>>5) + (threadIdx.x>>5);
  int lane = threadIdx.x & 31;
  if (gid>=G) return;
  float s = 0.f;
  for (int i=lane; i<ENC; i+=32) s += out1[(size_t)gid*ENC+i]*l2W[i];
  #pragma unroll
  for (int off=16; off; off>>=1) s += __shfl_down(s, off, 32);
  if (lane==0) out[gid] = sigmoidf_(s + l2b[0]);
}

// =====================================================================
extern "C" void kernel_launch(void* const* d_in, const int* in_sizes, int n_in,
                              void* d_out, int out_size, void* d_ws, size_t ws_size,
                              hipStream_t stream){
  const int N   = in_sizes[0]/9;     // 262144
  const int E   = in_sizes[1]/2;     // 524288
  const int G   = 8192;
  const int ENC = 128;

  const int*   x     = (const int*)d_in[0];
  const int*   ei    = (const int*)d_in[1];
  const int*   batch = (const int*)d_in[2];
  const float* emb   = (const float*)d_in[4];
  const float* W1=(const float*)d_in[5],  *b1=(const float*)d_in[6];
  const float* W2=(const float*)d_in[7],  *b2=(const float*)d_in[8];
  const float* W3=(const float*)d_in[9],  *b3=(const float*)d_in[10];
  const float* gamma=(const float*)d_in[11], *beta=(const float*)d_in[12];
  const float* Wi=(const float*)d_in[13], *Wh=(const float*)d_in[14];
  const float* bi=(const float*)d_in[15], *bh=(const float*)d_in[16];
  const float* l1W=(const float*)d_in[17], *l1b=(const float*)d_in[18];
  const float* l2W=(const float*)d_in[19], *l2b=(const float*)d_in[20];
  float* out = (float*)d_out;

  // ---- workspace layout (float units) ----
  float* ws = (float*)d_ws;
  const size_t NH = (size_t)N*HD;
  float*  bufA  = ws;                          // node features (ping)     [NH]
  float*  bufB  = ws + NH;                     // node features (pong) / set2set [NH]
  __bf16* abf   = (__bf16*)(ws + 2*NH);        // bf16 activation staging  [NH bf16]
  __bf16* wcbf  = (__bf16*)(ws + 2*NH + NH/2); // conv weight bf16 (HD*HD)
  float*  dinv  = ws + 2*NH + NH/2 + (HD*HD)/2;  // [N]
  float*  stats = dinv + N;                      // [2*HD]

  // set2set carve (inside bufB, free after last conv)
  float*    qstar = bufB;                       // G*2H
  float*    hs    = qstar + (size_t)G*2*HD;     // G*H
  float*    cs    = hs    + (size_t)G*HD;       // G*H
  float*    gates = cs    + (size_t)G*HD;       // G*4H
  float*    e_arr = gates + (size_t)G*4*HD;     // N
  unsigned* m_arr = (unsigned*)(e_arr + N);     // G
  float*    z_arr = (float*)m_arr + G;          // G
  float*    r_arr = z_arr + G;                  // G*H
  float*    out1  = r_arr + (size_t)G*HD;       // G*ENC
  __bf16*   wibf  = (__bf16*)(out1 + (size_t)G*ENC);   // 4H*2H bf16
  __bf16*   whbf  = wibf + (size_t)4*HD*2*HD;          // 4H*H bf16
  __bf16*   l1bf  = whbf + (size_t)4*HD*HD;            // ENC*2H bf16
  __bf16*   qsbf  = l1bf + (size_t)ENC*2*HD;           // G*2H bf16
  __bf16*   hsbf  = qsbf + (size_t)G*2*HD;             // G*H bf16

  auto cdiv = [](int a, int b){ return (a+b-1)/b; };

  // 1) embedding + degrees
  embed_kernel<<<N, HD, 0, stream>>>(x, emb, bufA, N);
  fill_f32<<<cdiv(N,256), 256, 0, stream>>>(dinv, 0.f, N);
  degree_kernel<<<cdiv(E,256), 256, 0, stream>>>(ei, dinv, E);
  dinv_kernel<<<cdiv(N,256), 256, 0, stream>>>(dinv, N);

  // 2) three GCN convs (ping-pong bufA <-> bufB)
  const float* Wl[3] = {W1, W2, W3};
  const float* bl[3] = {b1, b2, b3};
  const int NH4 = (int)(NH/4);
  for (int layer=0; layer<3; ++layer){
    cvt_bf16x4_kernel<<<cdiv(HD*HD/4,256), 256, 0, stream>>>(Wl[layer], wcbf, HD*HD/4);
    cvt_bf16x4_kernel<<<cdiv(NH4,256), 256, 0, stream>>>(bufA, abf, NH4);
    // hw = h @ W^T   (abf -> bufB); wave computes 16x128, A read twice (L2-resident bf16)
    gemm_wmma<8,HD><<<dim3(N/16, HD/128), 32, 0, stream>>>(abf, wcbf, nullptr, bufB,
                                                           N, HD, 0);
    // agg -> bufA
    agg_init_kernel<<<N, HD, 0, stream>>>(bufB, dinv, bl[layer], bufA, N);
    edge_scatter_kernel<<<E, HD, 0, stream>>>(ei, bufB, dinv, bufA, E);
    if (layer < 2){
      fill_f32<<<cdiv(2*HD,256), 256, 0, stream>>>(stats, 0.f, 2*HD);
      relu_stats_kernel<<<cdiv(N,256), HD, 0, stream>>>(bufA, stats, N);
      bn_apply_kernel<<<N, HD, 0, stream>>>(bufA, stats, gamma, beta, N);
    }
  }
  // h_final = bufA; bufB free for set2set

  // 3) set2set (4 processing steps)
  cvt_bf16x4_kernel<<<cdiv(4*HD*2*HD/4,256), 256, 0, stream>>>(Wi,  wibf, 4*HD*2*HD/4);
  cvt_bf16x4_kernel<<<cdiv(4*HD*HD/4,256),   256, 0, stream>>>(Wh,  whbf, 4*HD*HD/4);
  cvt_bf16x4_kernel<<<cdiv(ENC*2*HD/4,256),  256, 0, stream>>>(l1W, l1bf, ENC*2*HD/4);
  fill_f32<<<cdiv(G*2*HD,256), 256, 0, stream>>>(qstar, 0.f, G*2*HD);
  fill_f32<<<cdiv(G*HD,256),   256, 0, stream>>>(hs,    0.f, G*HD);
  fill_f32<<<cdiv(G*HD,256),   256, 0, stream>>>(cs,    0.f, G*HD);
  for (int step=0; step<4; ++step){
    // gates = qstar @ Wi^T + bi  +  hs @ Wh^T + bh
    cvt_bf16x4_kernel<<<cdiv(G*2*HD/4,256), 256, 0, stream>>>(qstar, qsbf, G*2*HD/4);
    cvt_bf16x4_kernel<<<cdiv(G*HD/4,256),   256, 0, stream>>>(hs,    hsbf, G*HD/4);
    gemm_wmma<8,2*HD><<<dim3(G/16, (4*HD)/128), 32, 0, stream>>>(qsbf, wibf, bi, gates,
                                                                 G, 4*HD, 0);
    gemm_wmma<8,HD><<<dim3(G/16, (4*HD)/128), 32, 0, stream>>>(hsbf, whbf, bh, gates,
                                                               G, 4*HD, 1);
    lstm_kernel<<<cdiv(G*HD,256), 256, 0, stream>>>(gates, hs, cs, G*HD);
    // scatter softmax + readout
    fill_u32<<<cdiv(G,256), 256, 0, stream>>>(m_arr, 0x007FFFFFu, G);   // ord(-inf)
    fill_f32<<<cdiv(G,256), 256, 0, stream>>>(z_arr, 0.f, G);
    fill_f32<<<cdiv(G*HD,256), 256, 0, stream>>>(r_arr, 0.f, G*HD);
    e_dot_kernel<<<cdiv(N,4), 128, 0, stream>>>(bufA, hs, batch, e_arr, m_arr, N);
    zexp_kernel<<<cdiv(N,256), 256, 0, stream>>>(e_arr, m_arr, z_arr, batch, N);
    r_scatter_kernel<<<N, HD, 0, stream>>>(e_arr, z_arr, bufA, batch, r_arr);
    qstar_update_kernel<<<cdiv(G*HD,256), 256, 0, stream>>>(hs, r_arr, qstar, G*HD);
  }

  // 4) head: out1 = qstar @ lin1_W^T + lin1_b ; out = sigmoid(out1 @ lin2_W^T + lin2_b)
  cvt_bf16x4_kernel<<<cdiv(G*2*HD/4,256), 256, 0, stream>>>(qstar, qsbf, G*2*HD/4);
  gemm_wmma<8,2*HD><<<dim3(G/16, 1), 32, 0, stream>>>(qsbf, l1bf, l1b, out1,
                                                      G, ENC, 0);
  final_kernel<<<cdiv(G,8), 256, 0, stream>>>(out1, l2W, l2b, out, G, ENC);
}
